// Linear_36249523978774
// MI455X (gfx1250) — matrix operand order repaired and checked by reading, not compiled, not run on hardware
//
#include <hip/hip_runtime.h>

// ---------------------------------------------------------------------------
// int8-weight linear  (M=8192, N=11008, K=4096)  on MI455X / gfx1250
//   out[m,n] = (sum_k x[m,k] * w[n,k]) * scale[n]
// Compute-bound (AI ~1370 FLOP/B) -> bf16 WMMA, fp32 accumulation.
// v5: block tile 128x256 (8 waves, 64x64 per wave -> 16 WMMA/wave/iter,
//     2x compute density per staged byte); saddr addressing; int8->bf16 via
//     v_cvt_f32_ubyteN + packed-bf16 bias subtract; double-buffered LDS.
// ---------------------------------------------------------------------------

typedef __attribute__((ext_vector_type(16))) __bf16 v16bf;
typedef __attribute__((ext_vector_type(4)))  __bf16 v4bf;
typedef __attribute__((ext_vector_type(8)))  float  v8f;

#define M_TOTAL 8192
#define N_TOTAL 11008
#define K_TOTAL 4096

#define BM 128
#define BN 256
#define BK 32
#define NK (K_TOTAL / BK)   // 128
#define LDA 40   // BK + 8 bf16 pad: 80B row stride, keeps 16B alignment
#define LDB 40

// 4 packed signed bytes -> 4 bf16 (exact).
// s8 = (u8 ^ 0x80) - 128; uitofp(byte) -> v_cvt_f32_ubyte_i (1 op/byte);
// bias subtract in packed bf16 (exact for [0,255]) -> v_pk_add_bf16.
__device__ __forceinline__ uint2 s8x4_to_bf16x4(unsigned q) {
  const unsigned t = q ^ 0x80808080u;
  union { v4bf v; uint2 u; } pk;
  pk.v[0] = (__bf16)(float)(t & 0xffu);
  pk.v[1] = (__bf16)(float)((t >> 8)  & 0xffu);
  pk.v[2] = (__bf16)(float)((t >> 16) & 0xffu);
  pk.v[3] = (__bf16)(float)(t >> 24);
  pk.v = pk.v - (__bf16)128.0f;
  return pk.u;
}

__device__ __forceinline__ uint2 f32x4_to_bf16x4(float4 f) {
  union { __bf16 b[4]; uint2 u; } pk;
  pk.b[0] = (__bf16)f.x; pk.b[1] = (__bf16)f.y;  // v_cvt_pk_bf16_f32 pairs
  pk.b[2] = (__bf16)f.z; pk.b[3] = (__bf16)f.w;
  return pk.u;
}

__global__ __launch_bounds__(256)
void q8_linear_wmma_bf16(const float* __restrict__ x,
                         const signed char* __restrict__ w,
                         const float* __restrict__ scale,
                         float* __restrict__ out)
{
  __shared__ __bf16 As[2][BM * LDA];   // 2 x 10 KB
  __shared__ __bf16 Bs[2][BN * LDB];   // 2 x 20 KB

  const int tid   = threadIdx.x;
  const int lane  = tid & 31;
  const int wave  = tid >> 5;     // 0..7
  const int waveM = wave >> 2;    // 0..1 : 64 rows of C each
  const int waveN = wave & 3;     // 0..3 : 64 cols of C each

  const int bm = blockIdx.y * BM;
  const int bn = blockIdx.x * BN;

  // --- per-thread 32-bit element offsets (loop-invariant) ---
  int aOff[4], aLds[4];
  #pragma unroll
  for (int p = 0; p < 4; ++p) {
    const int id  = p * 256 + tid;
    const int row = id >> 3;          // 0..127
    const int col = (id & 7) << 2;    // 0,4,..,28
    aOff[p] = row * K_TOTAL + col;
    aLds[p] = row * LDA + col;
  }
  const int bOff = tid * K_TOTAL;     // one full 32B row of w per thread
  const int bLds = tid * LDB;

  const float*       x0 = x + (size_t)bm * K_TOTAL;   // uniform bases
  const signed char* w0 = w + (size_t)bn * K_TOTAL;

  v8f acc[4][4];
  {
    v8f z = {};
    #pragma unroll
    for (int i = 0; i < 4; ++i)
      #pragma unroll
      for (int j = 0; j < 4; ++j)
        acc[i][j] = z;
  }

  const int r15 = lane & 15;
  const int h8  = (lane >> 4) << 3;   // A: element offset 0 or 8
  const int h16 = (lane >> 4) << 4;   // unused: B also reads at h8 (offset 0 or 8)

  const int aFragBase = (waveM * 64 + r15) * LDA + h8;
  const int bFragBase = (waveN * 64 + r15) * LDB + h8;

  // ---- stage one B row (32 int8 -> 32 bf16) into a buffer ----
  auto stage_b = [&](int buf, int4 q0, int4 q1) {
    const uint2 c0 = s8x4_to_bf16x4((unsigned)q0.x);
    const uint2 c1 = s8x4_to_bf16x4((unsigned)q0.y);
    const uint2 c2 = s8x4_to_bf16x4((unsigned)q0.z);
    const uint2 c3 = s8x4_to_bf16x4((unsigned)q0.w);
    const uint2 c4 = s8x4_to_bf16x4((unsigned)q1.x);
    const uint2 c5 = s8x4_to_bf16x4((unsigned)q1.y);
    const uint2 c6 = s8x4_to_bf16x4((unsigned)q1.z);
    const uint2 c7 = s8x4_to_bf16x4((unsigned)q1.w);
    uint4* dst = (uint4*)&Bs[buf][bLds];
    dst[0] = make_uint4(c0.x, c0.y, c1.x, c1.y);
    dst[1] = make_uint4(c2.x, c2.y, c3.x, c3.y);
    dst[2] = make_uint4(c4.x, c4.y, c5.x, c5.y);
    dst[3] = make_uint4(c6.x, c6.y, c7.x, c7.y);
  };

  // ---- WMMA on one staged buffer: 4x4 fragments = 16 WMMAs ----
  auto compute_tile = [&](int cur) {
    v16bf afrag[4], bfrag[4];
    #pragma unroll
    for (int fm = 0; fm < 4; ++fm) {
      const __bf16* pa = &As[cur][aFragBase + fm * 16 * LDA];
      union { uint4 q[2]; v16bf v; } ua;
      ua.q[0] = *(const uint4*)(pa);        // K = h8..h8+7
      ua.q[1] = *(const uint4*)(pa + 16);   // K = h8+16..h8+23
      afrag[fm] = ua.v;
    }
    #pragma unroll
    for (int fn = 0; fn < 4; ++fn) {
      const __bf16* pb = &Bs[cur][bFragBase + fn * 16 * LDB];
      union { uint4 q[2]; v16bf v; } ub;
      ub.q[0] = *(const uint4*)(pb);        // K = h8..h8+7
      ub.q[1] = *(const uint4*)(pb + 16);   // K = h8+16..h8+23
      bfrag[fn] = ub.v;
    }
    #pragma unroll
    for (int fm = 0; fm < 4; ++fm)
      #pragma unroll
      for (int fn = 0; fn < 4; ++fn)
        acc[fm][fn] = __builtin_amdgcn_wmma_f32_16x16x32_bf16(
            false, afrag[fm], false, bfrag[fn],
            (short)0, acc[fm][fn], false, false);
  };

  // ---- prologue: stage tile 0 into buffer 0 ----
  {
    #pragma unroll
    for (int p = 0; p < 4; ++p) {
      const float4 f = *(const float4*)(x0 + aOff[p]);
      *(uint2*)&As[0][aLds[p]] = f32x4_to_bf16x4(f);
    }
    const int4 q0 = *(const int4*)(w0 + bOff);
    const int4 q1 = *(const int4*)(w0 + bOff + 16);
    stage_b(0, q0, q1);
  }
  __syncthreads();

  // ---- steady state: branch-free; stage kt+1 while computing kt ----
  for (int kt = 0; kt < NK - 1; ++kt) {
    const int cur = kt & 1;
    const int nxt = cur ^ 1;

    // uniform bases advance with K; per-lane offsets invariant -> saddr
    const float*       xIter = x0 + (kt + 1) * BK;
    const signed char* wIter = w0 + (kt + 1) * BK;

    // global loads for tile kt+1 issue first -> latency hides under WMMA
    float4 fA[4];
    #pragma unroll
    for (int p = 0; p < 4; ++p) {
      fA[p] = *(const float4*)(xIter + aOff[p]);
      if (p == 0) __builtin_prefetch(xIter + aOff[0] + BK, 0, 1);  // kt+2
    }
    const int4 qB0 = *(const int4*)(wIter + bOff);
    const int4 qB1 = *(const int4*)(wIter + bOff + 16);
    __builtin_prefetch(wIter + bOff + BK, 0, 1);

    compute_tile(cur);

    // convert + store tile kt+1 into the other buffer
    #pragma unroll
    for (int p = 0; p < 4; ++p)
      *(uint2*)&As[nxt][aLds[p]] = f32x4_to_bf16x4(fA[p]);
    stage_b(nxt, qB0, qB1);

    __syncthreads();
  }

  // ---- last tile: compute only ----
  compute_tile((NK - 1) & 1);

  // ---- epilogue: apply per-output-channel scale in fp32, store ----
  const int nBase = bn + waveN * 64 + r15;
  const int mOff  = (lane >> 4) << 3;   // VGPR r: M=r (lanes 0-15), M=r+8 (16-31)
  #pragma unroll
  for (int fn = 0; fn < 4; ++fn) {
    const int n = nBase + fn * 16;
    const float sc = scale[n];
    #pragma unroll
    for (int fm = 0; fm < 4; ++fm) {
      const int mBase = bm + waveM * 64 + fm * 16 + mOff;
      #pragma unroll
      for (int r = 0; r < 8; ++r)
        out[(size_t)(mBase + r) * N_TOTAL + n] = acc[fm][fn][r] * sc;
    }
  }
}

extern "C" void kernel_launch(void* const* d_in, const int* in_sizes, int n_in,
                              void* d_out, int out_size, void* d_ws, size_t ws_size,
                              hipStream_t stream) {
  const float*       x  = (const float*)d_in[0];
  const signed char* w  = (const signed char*)d_in[1];   // int8 weight
  const float*       sc = (const float*)d_in[2];
  float*             o  = (float*)d_out;

  dim3 grid(N_TOTAL / BN, M_TOTAL / BM);   // (43, 64)
  q8_linear_wmma_bf16<<<grid, 256, 0, stream>>>(x, w, sc, o);
}
